// SelfAttention_84602265796637
// MI455X (gfx1250) — matrix operand order repaired and checked by reading, not compiled, not run on hardware
//
#include <hip/hip_runtime.h>
#include <hip/hip_bf16.h>

typedef unsigned short u16;
typedef __attribute__((ext_vector_type(16))) __bf16 v16bf;
typedef __attribute__((ext_vector_type(8)))  __bf16 v8bf;
typedef __attribute__((ext_vector_type(8)))  float  v8f;

#define BB   8
#define NN   2048
#define DIN  512
#define DATT 512
#define BSTR 40          // padded LDS column stride in elements (80B: conflict-free b128 reads)

#if defined(__has_builtin)
#  if __has_builtin(__builtin_amdgcn_global_load_async_to_lds_b128)
#    define USE_ASYNC_LDS 1
#  endif
#endif
#ifndef USE_ASYNC_LDS
#  define USE_ASYNC_LDS 0
#endif

#if USE_ASYNC_LDS
// Builtin signature (from hipcc diagnostic): (int4 __device__*, int4 __shared__*, Ii offset, Ii cpol)
typedef int v4i_vs __attribute__((vector_size(16)));
typedef __attribute__((address_space(1))) v4i_vs as1_v4i;
typedef __attribute__((address_space(3))) v4i_vs as3_v4i;
#endif

// ---------- helpers ----------

__device__ __forceinline__ u16 f2bf(float f) {
    unsigned u = __float_as_uint(f);
    u += 0x7fffu + ((u >> 16) & 1u);       // round-to-nearest-even
    return (u16)(u >> 16);
}

union BFrag { v16bf v; v8bf h[2]; };

// A-matrix 16x32 bf16 fragment: lane L holds row m=L&15,
// K in [half*16, half*16+16) contiguous  (half = L>>4).
__device__ __forceinline__ v16bf load_afrag(const u16* __restrict__ rowPtr, int k, int half) {
    BFrag f;
    f.h[0] = *reinterpret_cast<const v8bf*>(rowPtr + k + half * 16);
    f.h[1] = *reinterpret_cast<const v8bf*>(rowPtr + k + half * 16 + 8);
    return f.v;
}

// B-matrix 32x16 bf16 fragment from an LDS-staged column (contiguous 32 k-elems).
__device__ __forceinline__ v16bf load_bfrag_lds(const u16* colPtr, int half) {
    BFrag f;
    const u16* p = colPtr + half * 16;
    f.h[0] = *reinterpret_cast<const v8bf*>(p);
    f.h[1] = *reinterpret_cast<const v8bf*>(p + 8);
    return f.v;
}

__device__ __forceinline__ v8f wmma_bf16(v16bf a, v16bf b, v8f c) {
    return __builtin_amdgcn_wmma_f32_16x16x32_bf16(false, a, false, b, (short)0, c, false, false);
}

template<int N>
__device__ __forceinline__ void wait_async() {
#if USE_ASYNC_LDS
#  if __has_builtin(__builtin_amdgcn_s_wait_asynccnt)
    __builtin_amdgcn_s_wait_asynccnt(N);
#  else
    asm volatile("s_wait_asynccnt %0" :: "i"(N) : "memory");
#  endif
#endif
}

// Stage one 64-col x 32-k bf16 B tile into LDS (4KB): 256 transfers of 16B, one per thread.
__device__ __forceinline__ void fill_b(u16* dst, const u16* bCols, size_t ldb, int kOff, int tid) {
    const int col = tid >> 2;          // 0..63
    const int part = tid & 3;          // 0..3 (8 elems = 16B each)
    const u16* src = bCols + (size_t)col * ldb + kOff + part * 8;
    u16* d = dst + col * BSTR + part * 8;
#if USE_ASYNC_LDS
    __builtin_amdgcn_global_load_async_to_lds_b128((as1_v4i*)src, (as3_v4i*)d, 0, 0);
#else
    *reinterpret_cast<v8bf*>(d) = *reinterpret_cast<const v8bf*>(src);
#endif
}

// Block-tiled GEMM mainloop: 8 waves x (16x64) = 128x64 tile; B staged via LDS.
// aRow: global ptr to this lane's A row (row = mBase + (lane&15)), contiguous K.
// bCols: global ptr to column 0 of the 64-wide B tile (each column contiguous K, stride ldb).
template<int KTOT>
__device__ __forceinline__ void gemm_mainloop(const u16* __restrict__ aRow,
                                              const u16* __restrict__ bCols, size_t ldb,
                                              u16* sB, int tid, v8f* acc) {
    const int lane = tid & 31;
    const int half = lane >> 4;
    const int l15  = lane & 15;
    constexpr int NK = KTOT / 32;

    auto compute = [&](int k, const u16* buf) {
        __builtin_prefetch(aRow + k + 64, 0, 3);
        v16bf a = load_afrag(aRow, k, half);
        acc[0] = wmma_bf16(a, load_bfrag_lds(buf + ( 0 + l15) * BSTR, half), acc[0]);
        acc[1] = wmma_bf16(a, load_bfrag_lds(buf + (16 + l15) * BSTR, half), acc[1]);
        acc[2] = wmma_bf16(a, load_bfrag_lds(buf + (32 + l15) * BSTR, half), acc[2]);
        acc[3] = wmma_bf16(a, load_bfrag_lds(buf + (48 + l15) * BSTR, half), acc[3]);
    };

#if USE_ASYNC_LDS
    fill_b(sB, bCols, ldb, 0, tid);                       // prologue: buffer 0
#pragma unroll 2
    for (int kc = 0; kc < NK - 1; ++kc) {
        u16* cur = sB + (kc & 1) * (64 * BSTR);
        u16* nxt = sB + ((kc + 1) & 1) * (64 * BSTR);
        fill_b(nxt, bCols, ldb, (kc + 1) * 32, tid);      // ASYNCcnt += 1 (per wave)
        wait_async<1>();                                  // this wave's cur-fill complete
        __syncthreads();                                  // everyone's cur-fill complete
        compute(kc * 32, cur);
        __syncthreads();                                  // cur fully consumed before reuse
    }
    wait_async<0>();
    __syncthreads();
    compute((NK - 1) * 32, sB + ((NK - 1) & 1) * (64 * BSTR));
#else
#pragma unroll 2
    for (int kc = 0; kc < NK; ++kc) {
        __syncthreads();                                  // prior compute done with sB
        fill_b(sB, bCols, ldb, kc * 32, tid);             // load + ds_store
        __syncthreads();
        compute(kc * 32, sB);
    }
#endif
}

// ---------- conversion kernels ----------

__global__ void __launch_bounds__(256) k_convert_x(const float* __restrict__ x,
                                                   u16* __restrict__ xb) {
    int t = blockIdx.x * 256 + threadIdx.x;               // one float4 per thread
    float4 v = reinterpret_cast<const float4*>(x)[t];
    unsigned lo = (unsigned)f2bf(v.x) | ((unsigned)f2bf(v.y) << 16);
    unsigned hi = (unsigned)f2bf(v.z) | ((unsigned)f2bf(v.w) << 16);
    reinterpret_cast<uint2*>(xb)[t] = make_uint2(lo, hi);
}

// Wt[z][n][k] = W_z[k][n]  (column of W becomes contiguous row of Wt)
__global__ void __launch_bounds__(256) k_convert_w(const float* __restrict__ Wq,
                                                   const float* __restrict__ Wk,
                                                   const float* __restrict__ Wv,
                                                   u16* __restrict__ Wt3) {
    int idx = blockIdx.x * 256 + threadIdx.x;             // 0 .. 512*512-1
    int z   = blockIdx.y;                                 // 0..2
    const float* W = (z == 0) ? Wq : (z == 1) ? Wk : Wv;
    int n = idx >> 9;
    int k = idx & 511;
    Wt3[(size_t)z * DIN * DATT + idx] = f2bf(W[k * DATT + n]);
}

// ---------- QKV projection: [N,512] = x @ W + b ----------

__global__ void __launch_bounds__(256) k_qkv(const u16* __restrict__ xb,
                                             const u16* __restrict__ Wt3,
                                             const float* __restrict__ bq,
                                             const float* __restrict__ bk,
                                             const float* __restrict__ bv,
                                             u16* __restrict__ Qb,
                                             u16* __restrict__ Kb,
                                             u16* __restrict__ Vt) {
    __shared__ __align__(16) u16 sB[2 * 64 * BSTR];
    const int tid  = threadIdx.x;
    const int lane = tid & 31;
    const int wave = tid >> 5;
    const int half = lane >> 4;
    const int l15  = lane & 15;
    const int z = blockIdx.z;
    const int b = z / 3, which = z % 3;
    const int nBase = blockIdx.x * 64;
    const int mBase = blockIdx.y * 128 + wave * 16;

    const u16* aRow  = xb + (size_t)b * NN * DIN + (size_t)(mBase + l15) * DIN;
    const u16* bCols = Wt3 + (size_t)which * DIN * DATT + (size_t)nBase * DIN;

    v8f zv = {0,0,0,0,0,0,0,0};
    v8f acc[4] = {zv, zv, zv, zv};
    gemm_mainloop<DIN>(aRow, bCols, DIN, sB, tid, acc);

    const float* bias = (which == 0) ? bq : (which == 1) ? bk : bv;
    const float b0 = bias[nBase +  0 + l15];
    const float b1 = bias[nBase + 16 + l15];
    const float b2 = bias[nBase + 32 + l15];
    const float b3 = bias[nBase + 48 + l15];
#pragma unroll
    for (int r = 0; r < 8; ++r) { acc[0][r] += b0; acc[1][r] += b1; acc[2][r] += b2; acc[3][r] += b3; }

    if (which < 2) {                                      // Q, K row-major [N, DATT]
        u16* dst = ((which == 0) ? Qb : Kb) + (size_t)b * NN * DATT;
#pragma unroll
        for (int r = 0; r < 8; ++r) {
            size_t row = (size_t)(mBase + half * 8 + r) * DATT + nBase + l15;
            dst[row +  0] = f2bf(acc[0][r]);
            dst[row + 16] = f2bf(acc[1][r]);
            dst[row + 32] = f2bf(acc[2][r]);
            dst[row + 48] = f2bf(acc[3][r]);
        }
    } else {                                              // V transposed: Vt[e][j]
        u16* dst = Vt + (size_t)b * DATT * NN;
#pragma unroll
        for (int r = 0; r < 8; ++r) {
            int j = mBase + half * 8 + r;
            dst[(size_t)(nBase +  0 + l15) * NN + j] = f2bf(acc[0][r]);
            dst[(size_t)(nBase + 16 + l15) * NN + j] = f2bf(acc[1][r]);
            dst[(size_t)(nBase + 32 + l15) * NN + j] = f2bf(acc[2][r]);
            dst[(size_t)(nBase + 48 + l15) * NN + j] = f2bf(acc[3][r]);
        }
    }
}

// ---------- scores: S[b,i,j] = scale * Q[b,i,:].K[b,j,:]  (fp32 into d_out weights region) ----------

__global__ void __launch_bounds__(256) k_scores(const u16* __restrict__ Qb,
                                                const u16* __restrict__ Kb,
                                                float* __restrict__ Sw) {
    __shared__ __align__(16) u16 sB[2 * 64 * BSTR];
    const int tid  = threadIdx.x;
    const int lane = tid & 31;
    const int wave = tid >> 5;
    const int half = lane >> 4;
    const int l15  = lane & 15;
    const int b = blockIdx.z;
    const int jBase = blockIdx.x * 64;
    const int iBase = blockIdx.y * 128 + wave * 16;

    const u16* aRow  = Qb + (size_t)b * NN * DATT + (size_t)(iBase + l15) * DATT;
    const u16* bCols = Kb + (size_t)b * NN * DATT + (size_t)jBase * DATT;

    v8f zv = {0,0,0,0,0,0,0,0};
    v8f acc[4] = {zv, zv, zv, zv};
    gemm_mainloop<DATT>(aRow, bCols, DATT, sB, tid, acc);

    const float scale = 0.044194173824159216f;            // 1/sqrt(512)
    float* S = Sw + (size_t)b * NN * NN;
#pragma unroll
    for (int r = 0; r < 8; ++r) {
        size_t row = (size_t)(iBase + half * 8 + r) * NN + jBase + l15;
        S[row +  0] = acc[0][r] * scale;
        S[row + 16] = acc[1][r] * scale;
        S[row + 32] = acc[2][r] * scale;
        S[row + 48] = acc[3][r] * scale;
    }
}

// ---------- column softmax reduction over i (axis=1): per (b,j) streaming max+sum ----------

__global__ void __launch_bounds__(256) k_colreduce(const float* __restrict__ Sw,
                                                   float* __restrict__ cmax,
                                                   float* __restrict__ csum) {
    const int b = blockIdx.y;
    const int j = blockIdx.x * 256 + threadIdx.x;
    const float* p = Sw + (size_t)b * NN * NN + j;
    float m = -3.4e38f, l = 0.f;
    for (int i = 0; i < NN; ++i) {
        float s = p[(size_t)i * NN];
        float mn = fmaxf(m, s);
        l = l * __expf(m - mn) + __expf(s - mn);
        m = mn;
    }
    cmax[b * NN + j] = m;
    csum[b * NN + j] = l;
}

// ---------- normalize in place (fp32 output) + bf16 copy for PV GEMM ----------

__global__ void __launch_bounds__(256) k_norm(float* __restrict__ Sw,
                                              u16* __restrict__ Pb,
                                              const float* __restrict__ cmax,
                                              const float* __restrict__ csum) {
    size_t idx = (size_t)blockIdx.x * 256 + threadIdx.x;  // 0 .. B*N*N-1
    int j = (int)(idx & (NN - 1));
    int b = (int)(idx >> 22);                             // N*N = 2^22
    float m    = cmax[b * NN + j];
    float invl = 1.0f / csum[b * NN + j];
    float w = __expf(Sw[idx] - m) * invl;
    Sw[idx] = w;
    Pb[idx] = f2bf(w);
}

// ---------- out = P @ V  via Vt ----------

__global__ void __launch_bounds__(256) k_av(const u16* __restrict__ Pb,
                                            const u16* __restrict__ Vt,
                                            float* __restrict__ outp) {
    __shared__ __align__(16) u16 sB[2 * 64 * BSTR];
    const int tid  = threadIdx.x;
    const int lane = tid & 31;
    const int wave = tid >> 5;
    const int half = lane >> 4;
    const int l15  = lane & 15;
    const int b = blockIdx.z;
    const int eBase = blockIdx.x * 64;
    const int iBase = blockIdx.y * 128 + wave * 16;

    const u16* aRow  = Pb + (size_t)b * NN * NN + (size_t)(iBase + l15) * NN;
    const u16* bCols = Vt + (size_t)b * DATT * NN + (size_t)eBase * NN;

    v8f zv = {0,0,0,0,0,0,0,0};
    v8f acc[4] = {zv, zv, zv, zv};
    gemm_mainloop<NN>(aRow, bCols, NN, sB, tid, acc);

    float* O = outp + (size_t)b * NN * DATT;
#pragma unroll
    for (int r = 0; r < 8; ++r) {
        size_t row = (size_t)(iBase + half * 8 + r) * DATT + eBase + l15;
        O[row +  0] = acc[0][r];
        O[row + 16] = acc[1][r];
        O[row + 32] = acc[2][r];
        O[row + 48] = acc[3][r];
    }
}

// ---------- launch ----------

extern "C" void kernel_launch(void* const* d_in, const int* in_sizes, int n_in,
                              void* d_out, int out_size, void* d_ws, size_t ws_size,
                              hipStream_t stream) {
    (void)in_sizes; (void)n_in; (void)out_size; (void)ws_size;

    const float* x  = (const float*)d_in[0];
    const float* Wq = (const float*)d_in[1];
    const float* bq = (const float*)d_in[2];
    const float* Wk = (const float*)d_in[3];
    const float* bk = (const float*)d_in[4];
    const float* Wv = (const float*)d_in[5];
    const float* bv = (const float*)d_in[6];

    float* outp = (float*)d_out;
    float* Sw   = outp + (size_t)BB * NN * DATT;          // weights region of d_out

    char* w = (char*)d_ws;
    u16* xb  = (u16*)w;  w += (size_t)BB * NN * DIN  * sizeof(u16);   // 16 MB
    u16* Wt3 = (u16*)w;  w += (size_t)3  * DIN * DATT * sizeof(u16);  // 1.5 MB
    u16* Qb  = (u16*)w;  w += (size_t)BB * NN * DATT * sizeof(u16);   // 16 MB
    u16* Kb  = (u16*)w;  w += (size_t)BB * NN * DATT * sizeof(u16);   // 16 MB
    u16* Vt  = (u16*)w;  w += (size_t)BB * DATT * NN * sizeof(u16);   // 16 MB
    u16* Pb  = (u16*)w;  w += (size_t)BB * NN * NN   * sizeof(u16);   // 64 MB
    float* cmax = (float*)w; w += (size_t)BB * NN * sizeof(float);
    float* csum = (float*)w;

    k_convert_x<<<(BB * NN * DIN / 4) / 256, 256, 0, stream>>>(x, xb);
    k_convert_w<<<dim3((DIN * DATT) / 256, 3), 256, 0, stream>>>(Wq, Wk, Wv, Wt3);
    k_qkv<<<dim3(DATT / 64, NN / 128, BB * 3), 256, 0, stream>>>(xb, Wt3, bq, bk, bv, Qb, Kb, Vt);
    k_scores<<<dim3(NN / 64, NN / 128, BB), 256, 0, stream>>>(Qb, Kb, Sw);
    k_colreduce<<<dim3(NN / 256, BB), 256, 0, stream>>>(Sw, cmax, csum);
    k_norm<<<(unsigned)((size_t)BB * NN * NN / 256), 256, 0, stream>>>(Sw, Pb, cmax, csum);
    k_av<<<dim3(DATT / 64, NN / 128, BB), 256, 0, stream>>>(Pb, Vt, outp);
}
